// Prompt_generator_8607114461813
// MI455X (gfx1250) — matrix operand order repaired and checked by reading, not compiled, not run on hardware
//
#include <hip/hip_runtime.h>

typedef __attribute__((ext_vector_type(16))) _Float16 v16h;
typedef __attribute__((ext_vector_type(8)))  _Float16 v8h;
typedef __attribute__((ext_vector_type(8)))  float    v8f;

#define NB   1024
#define NA   64
#define NG   31
#define NW   8
#define NH   512
#define NL   2
#define NS   32
#define NDK  32
#define K3H  1536
#define NT   256

// ---- workspace layout (_Float16 element offsets) ----
#define OFF_LR   0
#define SZ_LR    (K3H*NH)              // 786432
#define SZ_QKV1  (NH*NDK)              // 16384
#define OFF_Q    (OFF_LR + SZ_LR)
#define OFF_K    (OFF_Q + 2*SZ_QKV1)
#define OFF_V    (OFF_K + 2*SZ_QKV1)
#define SZ_D1    (NDK*NH)
#define OFF_D    (OFF_V + 2*SZ_QKV1)
#define OFF_LIN  (OFF_D + 2*SZ_D1)
#define SZ_LIN   (NH*NH)

#define SMEM_BYTES 218368

// ---------------------------------------------------------------- helpers
__device__ __forceinline__ v8f wmma16(v16h a, v16h b, v8f c) {
  return __builtin_amdgcn_wmma_f32_16x16x32_f16(false, a, false, b, (short)0, c,
                                                false, false);
}

// A-matrix 16x32 f16 fragment from row-major f16 array (ld = row stride elems).
// Lane L -> m=L%16, elements hold K = kh+{0..15}, kh = (L>=16)*16
// -> 16 contiguous f16 (32 B) per lane, matching the B fragments.
__device__ __forceinline__ v16h frag_a(const _Float16* base, int ld, int row0,
                                       int k0, int lane) {
  const int m  = lane & 15;
  const int kh = (lane >> 4) << 4;
  const _Float16* p = base + (size_t)(row0 + m) * ld + k0 + kh;
  v8h lo = *(const v8h*)(p);
  v8h hi = *(const v8h*)(p + 8);
  v16h r;
#pragma unroll
  for (int i = 0; i < 8; ++i) { r[i] = lo[i]; r[i + 8] = hi[i]; }
  return r;
}

// B-matrix fragment: lane L holds column n = n0 + L%16, K = kb+{0..15},
// kb = (L>=16)*16 -> 16 contiguous f16 (32 B) per lane.
__device__ __forceinline__ v16h frag_b16(const _Float16* p) {
  v8h lo = *(const v8h*)(p);
  v8h hi = *(const v8h*)(p + 8);
  v16h r;
#pragma unroll
  for (int i = 0; i < 8; ++i) { r[i] = lo[i]; r[i + 8] = hi[i]; }
  return r;
}

// packed weights: pack[(k/32)*N + n][k%32]
__device__ __forceinline__ v16h frag_b_pack(const _Float16* pack, int N, int kt,
                                            int n0, int lane) {
  const int n  = n0 + (lane & 15);
  const int kh = (lane >> 4) << 4;
  return frag_b16(pack + ((size_t)kt * N + n) * 32 + kh);
}

// B from row-major [n][k] storage (e.g. K matrix / transposed V in LDS)
__device__ __forceinline__ v16h frag_b_rows(const _Float16* base, int ld,
                                            int n0, int k0, int lane) {
  const int n  = n0 + (lane & 15);
  const int kh = (lane >> 4) << 4;
  return frag_b16(base + (size_t)n * ld + k0 + kh);
}

// ---------------------------------------------------------------- pack
__global__ void pack_w(const float* __restrict__ src, _Float16* __restrict__ dst,
                       int K, int N) {
  int idx = blockIdx.x * blockDim.x + threadIdx.x;
  if (idx >= K * N) return;
  int k = idx / N, n = idx - k * N;
  dst[((size_t)(k >> 5) * N + n) * 32 + (k & 31)] = (_Float16)src[idx];
}

// ---------------------------------------------------------------- fused main
__global__ __launch_bounds__(NT) void fused_kernel(
    const float* __restrict__ f_atom, const float* __restrict__ f_group,
    const float* __restrict__ fg, const float* __restrict__ alpha,
    const float* __restrict__ lr_b, const float* __restrict__ wq_b,
    const float* __restrict__ wk_b, const float* __restrict__ wv_b,
    const float* __restrict__ dense_b, const float* __restrict__ ln_g,
    const float* __restrict__ ln_b, const float* __restrict__ lin_b,
    const float* __restrict__ norm_g, const float* __restrict__ norm_b,
    const long long* __restrict__ mapping, const _Float16* __restrict__ ws,
    float* __restrict__ d_out) {
  extern __shared__ char smem[];
  _Float16* brics = (_Float16*)(smem + 0);            // 32 x 1536 f16
  float*    dh    = (float*)(smem + 0);               // 32 x 512 f32 (reuse)
  float*    hs    = (float*)(smem + 98304);           // 32 x 512 f32
  _Float16* hs16  = (_Float16*)(smem + 163840);       // 32 x 512 f16
  _Float16* q16   = (_Float16*)(smem + 196608);       // 32 x 32
  _Float16* k16   = (_Float16*)(smem + 198656);       // 32 x 32
  _Float16* vT16  = (_Float16*)(smem + 200704);       // 32 x 32 (transposed)
  _Float16* at16  = (_Float16*)(smem + 202752);       // 32 x 32
  _Float16* x16   = (_Float16*)(smem + 204800);       // 32 x 32
  float*    scor  = (float*)(smem + 206848);          // 32 x 32 f32
  unsigned* pmb   = (unsigned*)(smem + 210944);       // 32 x 16 bitmask
  unsigned char* maskb = (unsigned char*)(smem + 212992); // 32 x 32
  float*    red   = (float*)(smem + 214016);          // 512 f32
  float*    stats = (float*)(smem + 216064);          // 64 f32
  float*    fout  = (float*)(smem + 216320);          // 512 f32

  const int b    = blockIdx.x;
  const int tid  = threadIdx.x;
  const int wave = tid >> 5;
  const int lane = tid & 31;

  const _Float16* packLR  = ws + OFF_LR;
  const _Float16* packLin = ws + OFF_LIN;

  // ---------------- Phase 1: gather + build brics (f16, row-major) ----------
  for (int i = tid; i < K3H; i += NT) brics[i] = (_Float16)fg[i];
  for (int i = tid; i < NS * 16; i += NT) pmb[i] = 0u;

  const long long* map = mapping + (size_t)b * NG * NW;
  for (int g = 0; g < NG; ++g) {
    long long idx[NW];
#pragma unroll
    for (int w = 0; w < NW; ++w) idx[w] = map[g * NW + w];
    for (int h = tid; h < NH; h += NT) {
      float sm = 0.f, mx = -3.4e38f;
#pragma unroll
      for (int w = 0; w < NW; ++w) {
        float v = (idx[w] == 0)
                      ? 0.f
                      : f_atom[((size_t)b * NA + (size_t)(idx[w] - 1)) * NH + h];
        sm += v;
        mx = fmaxf(mx, v);
      }
      _Float16* row = brics + (size_t)(1 + g) * K3H;
      row[h]            = (_Float16)sm;
      row[NH + h]       = (_Float16)mx;
      row[2 * NH + h]   = (_Float16)f_group[((size_t)b * NG + g) * NH + h];
    }
  }
  __syncthreads();

  // ---------------- Phase 2: f_frag = relu(brics @ lr_w + lr_b) -------------
  {
    v8f acc[2][4] = {};
    for (int kt = 0; kt < K3H / 32; ++kt) {
      v16h a0 = frag_a(brics, K3H, 0, kt * 32, lane);
      v16h a1 = frag_a(brics, K3H, 16, kt * 32, lane);
#pragma unroll
      for (int ci = 0; ci < 4; ++ci) {
        v16h bf = frag_b_pack(packLR, NH, kt, (wave * 4 + ci) * 16, lane);
        acc[0][ci] = wmma16(a0, bf, acc[0][ci]);
        acc[1][ci] = wmma16(a1, bf, acc[1][ci]);
      }
    }
    const int mlo = (lane >> 4) << 3;
    const int nl  = lane & 15;
#pragma unroll
    for (int rt = 0; rt < 2; ++rt)
#pragma unroll
      for (int ci = 0; ci < 4; ++ci) {
        int n = (wave * 4 + ci) * 16 + nl;
        float bias = lr_b[n];
#pragma unroll
        for (int r = 0; r < 8; ++r) {
          int m = rt * 16 + mlo + r;
          float v = fmaxf(acc[rt][ci][r] + bias, 0.f);
          hs[m * NH + n]   = v;
          hs16[m * NH + n] = (_Float16)v;
          if (v != 0.f) atomicOr(&pmb[m * 16 + (n >> 5)], 1u << (n & 31));
        }
      }
  }
  __syncthreads();

  // mask[s][t] = (pm[s] . pm[t]) != 0
  for (int p = tid; p < NS * NS; p += NT) {
    int s = p >> 5, t = p & 31;
    unsigned any = 0;
#pragma unroll
    for (int i = 0; i < 16; ++i) any |= (pmb[s * 16 + i] & pmb[t * 16 + i]);
    maskb[p] = any ? 1 : 0;
  }
  __syncthreads();

  // ---------------- Phase 3: transformer layers -----------------------------
  for (int l = 0; l < NL; ++l) {
    // q/k/v : (32x512)@(512x32), 12 tile jobs over 8 waves
    for (int job = wave; job < 12; job += 8) {
      const int mat = job >> 2, rt = (job >> 1) & 1, ct = job & 1;
      const _Float16* wp =
          (mat == 0 ? ws + OFF_Q : mat == 1 ? ws + OFF_K : ws + OFF_V) +
          (size_t)l * SZ_QKV1;
      const float* bias =
          (mat == 0 ? wq_b : mat == 1 ? wk_b : wv_b) + l * NDK;
      v8f c = {};
      for (int kt = 0; kt < NH / 32; ++kt) {
        v16h a  = frag_a(hs16, NH, rt * 16, kt * 32, lane);
        v16h bf = frag_b_pack(wp, NDK, kt, ct * 16, lane);
        c = wmma16(a, bf, c);
      }
      int n = ct * 16 + (lane & 15);
      float bv = bias[n];
      const int mlo = (lane >> 4) << 3;
#pragma unroll
      for (int r = 0; r < 8; ++r) {
        int m = rt * 16 + mlo + r;
        float v = c[r] + bv;
        if (mat == 0)      q16[m * NDK + n] = (_Float16)v;
        else if (mat == 1) k16[m * NDK + n] = (_Float16)v;
        else               vT16[n * NDK + m] = (_Float16)v;  // transposed
      }
    }
    __syncthreads();

    // scores = q @ k^T * scale   (K = 32, single WMMA per tile)
    for (int job = wave; job < 4; job += 8) {
      const int rt = job >> 1, ct = job & 1;
      v16h a  = frag_a(q16, NDK, rt * 16, 0, lane);
      v16h bf = frag_b_rows(k16, NDK, ct * 16, 0, lane);
      v8f c = {};
      c = wmma16(a, bf, c);
      int t = ct * 16 + (lane & 15);
      const int mlo = (lane >> 4) << 3;
#pragma unroll
      for (int r = 0; r < 8; ++r)
        scor[(rt * 16 + mlo + r) * NS + t] = c[r] * 0.17677669529663687f;
    }
    __syncthreads();

    // masked softmax over t (fp32); row per thread
    if (tid < NS) {
      const int s = tid;
      float sc[NS];
      float mx = -3.4e38f;
#pragma unroll
      for (int t = 0; t < NS; ++t) {
        float v = maskb[s * NS + t] ? scor[s * NS + t] : -1.0e9f;
        sc[t] = v;
        mx = fmaxf(mx, v);
      }
      float sum = 0.f;
#pragma unroll
      for (int t = 0; t < NS; ++t) { sc[t] = __expf(sc[t] - mx); sum += sc[t]; }
      float inv = 1.f / sum;
#pragma unroll
      for (int t = 0; t < NS; ++t) {
        float a = sc[t] * inv;
        at16[s * NS + t] = (_Float16)a;
        if (l == NL - 1)
          d_out[(size_t)NB * NH + (size_t)b * NS * NS + s * NS + t] = a;
      }
    }
    __syncthreads();

    // x = attn @ v  (K = 32)
    for (int job = wave; job < 4; job += 8) {
      const int rt = job >> 1, ct = job & 1;
      v16h a  = frag_a(at16, NS, rt * 16, 0, lane);
      v16h bf = frag_b_rows(vT16, NS, ct * 16, 0, lane);
      v8f c = {};
      c = wmma16(a, bf, c);
      int n = ct * 16 + (lane & 15);
      const int mlo = (lane >> 4) << 3;
#pragma unroll
      for (int r = 0; r < 8; ++r)
        x16[(rt * 16 + mlo + r) * NDK + n] = (_Float16)c[r];
    }
    __syncthreads();

    // dh = x @ dense_w + dense_b + hs  (K = 32, 8 tiles / wave)
    {
      const _Float16* wp = ws + OFF_D + (size_t)l * SZ_D1;
      v16h a0 = frag_a(x16, NDK, 0, 0, lane);
      v16h a1 = frag_a(x16, NDK, 16, 0, lane);
      const int mlo = (lane >> 4) << 3;
      const int nl  = lane & 15;
#pragma unroll
      for (int ci = 0; ci < 4; ++ci) {
        int n0 = (wave * 4 + ci) * 16;
        v16h bf = frag_b_pack(wp, NH, 0, n0, lane);
        v8f c0 = {}, c1 = {};
        c0 = wmma16(a0, bf, c0);
        c1 = wmma16(a1, bf, c1);
        int n = n0 + nl;
        float bv = dense_b[l * NH + n];
#pragma unroll
        for (int r = 0; r < 8; ++r) {
          int m0 = mlo + r;
          dh[m0 * NH + n]        = c0[r] + bv + hs[m0 * NH + n];
          dh[(16 + m0) * NH + n] = c1[r] + bv + hs[(16 + m0) * NH + n];
        }
      }
    }
    __syncthreads();

    // layernorm(dh) -> hs, hs16  (eps 1e-6)
    {
      const int row = tid >> 3, sub = tid & 7;
      float s1 = 0.f, s2 = 0.f;
      for (int i = sub * 64; i < sub * 64 + 64; ++i) {
        float v = dh[row * NH + i];
        s1 += v; s2 += v * v;
      }
      red[(row * 8 + sub) * 2]     = s1;
      red[(row * 8 + sub) * 2 + 1] = s2;
      __syncthreads();
      if (sub == 0) {
        float S1 = 0.f, S2 = 0.f;
        for (int j = 0; j < 8; ++j) {
          S1 += red[(row * 8 + j) * 2];
          S2 += red[(row * 8 + j) * 2 + 1];
        }
        float mu = S1 / NH;
        stats[row * 2]     = mu;
        stats[row * 2 + 1] = rsqrtf(S2 / NH - mu * mu + 1e-6f);
      }
      __syncthreads();
      float mu = stats[row * 2], rs = stats[row * 2 + 1];
      for (int i = sub * 64; i < sub * 64 + 64; ++i) {
        float v = (dh[row * NH + i] - mu) * rs * ln_g[l * NH + i] + ln_b[l * NH + i];
        hs[row * NH + i]   = v;
        hs16[row * NH + i] = (_Float16)v;
      }
    }
    __syncthreads();
  }

  // ---------------- Phase 4: f_out row 0 ------------------------------------
  {
    v8f c[4] = {};
    for (int kt = 0; kt < NH / 32; ++kt) {
      v16h a = frag_a(hs16, NH, 0, kt * 32, lane);
#pragma unroll
      for (int ci = 0; ci < 4; ++ci) {
        v16h bf = frag_b_pack(packLin, NH, kt, (wave * 4 + ci) * 16, lane);
        c[ci] = wmma16(a, bf, c[ci]);
      }
    }
    if (lane < 16) {
#pragma unroll
      for (int ci = 0; ci < 4; ++ci) {
        int n = (wave * 4 + ci) * 16 + lane;   // row m=0 lives in VGPR0, lanes 0-15
        fout[n] = c[ci][0] + lin_b[n];
      }
    }
  }
  __syncthreads();

  // final layernorm (eps 1e-5) * alpha
  {
    float v0 = fout[tid], v1 = fout[tid + 256];
    red[tid]       = v0 + v1;
    red[256 + tid] = v0 * v0 + v1 * v1;
    __syncthreads();
    for (int off = 128; off > 0; off >>= 1) {
      if (tid < off) {
        red[tid]       += red[tid + off];
        red[256 + tid] += red[256 + tid + off];
      }
      __syncthreads();
    }
    if (tid == 0) {
      float mu = red[0] / NH;
      stats[0] = mu;
      stats[1] = rsqrtf(red[256] / NH - mu * mu + 1e-5f);
    }
    __syncthreads();
    float mu = stats[0], rs = stats[1], al = alpha[0];
    for (int i = tid; i < NH; i += NT)
      d_out[(size_t)b * NH + i] =
          ((fout[i] - mu) * rs * norm_g[i] + norm_b[i]) * al;
  }
}

// ---------------------------------------------------------------- launcher
extern "C" void kernel_launch(void* const* d_in, const int* in_sizes, int n_in,
                              void* d_out, int out_size, void* d_ws,
                              size_t ws_size, hipStream_t stream) {
  const float* f_atom  = (const float*)d_in[0];
  const float* f_group = (const float*)d_in[1];
  const float* fg      = (const float*)d_in[2];
  const float* alpha   = (const float*)d_in[3];
  const float* lr_w    = (const float*)d_in[4];
  const float* lr_b    = (const float*)d_in[5];
  const float* wq_w    = (const float*)d_in[6];
  const float* wq_b    = (const float*)d_in[7];
  const float* wk_w    = (const float*)d_in[8];
  const float* wk_b    = (const float*)d_in[9];
  const float* wv_w    = (const float*)d_in[10];
  const float* wv_b    = (const float*)d_in[11];
  const float* dense_w = (const float*)d_in[12];
  const float* dense_b = (const float*)d_in[13];
  const float* ln_g    = (const float*)d_in[14];
  const float* ln_b    = (const float*)d_in[15];
  const float* lin_w   = (const float*)d_in[16];
  const float* lin_b   = (const float*)d_in[17];
  const float* norm_g  = (const float*)d_in[18];
  const float* norm_b  = (const float*)d_in[19];
  const long long* mapping = (const long long*)d_in[20];
  _Float16* ws = (_Float16*)d_ws;
  float* out = (float*)d_out;

  auto grid = [](int total) { return (total + NT - 1) / NT; };

  pack_w<<<grid(K3H * NH), NT, 0, stream>>>(lr_w, ws + OFF_LR, K3H, NH);
  for (int l = 0; l < NL; ++l) {
    pack_w<<<grid(NH * NDK), NT, 0, stream>>>(wq_w + (size_t)l * NH * NDK,
                                              ws + OFF_Q + (size_t)l * SZ_QKV1,
                                              NH, NDK);
    pack_w<<<grid(NH * NDK), NT, 0, stream>>>(wk_w + (size_t)l * NH * NDK,
                                              ws + OFF_K + (size_t)l * SZ_QKV1,
                                              NH, NDK);
    pack_w<<<grid(NH * NDK), NT, 0, stream>>>(wv_w + (size_t)l * NH * NDK,
                                              ws + OFF_V + (size_t)l * SZ_QKV1,
                                              NH, NDK);
    pack_w<<<grid(NDK * NH), NT, 0, stream>>>(dense_w + (size_t)l * NDK * NH,
                                              ws + OFF_D + (size_t)l * SZ_D1,
                                              NDK, NH);
  }
  pack_w<<<grid(NH * NH), NT, 0, stream>>>(lin_w, ws + OFF_LIN, NH, NH);

  fused_kernel<<<NB, NT, SMEM_BYTES, stream>>>(
      f_atom, f_group, fg, alpha, lr_b, wq_b, wk_b, wv_b, dense_b, ln_g, ln_b,
      lin_b, norm_g, norm_b, mapping, ws, out);
}
